// Assign_841813590365
// MI455X (gfx1250) — matrix operand order repaired and checked by reading, not compiled, not run on hardware
//
#include <hip/hip_runtime.h>
#include <hip/hip_bf16.h>

// ---------------------------------------------------------------------------
// Assign kernel for MI455X (gfx1250, wave32, WMMA).
//   dot = seq @ C^T via v_wmma_f32_16x16x32_bf16 (cross term only in bf16;
//   x^2 and c^2 kept in f32, so score error << Gumbel noise sigma ~1.28).
//   HBM-bound: ~1.02 GB traffic -> ~44 us floor at 23.3 TB/s.
// ---------------------------------------------------------------------------

typedef __attribute__((ext_vector_type(16))) __bf16 v16bf;
typedef __attribute__((ext_vector_type(8)))  float  v8f;

union BFrag {
    v16bf v;
    unsigned int u[8];
};

// native f32 -> bf16 (RNE fptrunc; backend picks hardware cvt on gfx1250)
__device__ __forceinline__ void bf_pair(BFrag& f, int i, float lo, float hi) {
    f.v[2 * i]     = (__bf16)lo;
    f.v[2 * i + 1] = (__bf16)hi;
}

__device__ __forceinline__ float sumsq4(float4 a, float s) {
    s = fmaf(a.x, a.x, s); s = fmaf(a.y, a.y, s);
    s = fmaf(a.z, a.z, s); s = fmaf(a.w, a.w, s);
    return s;
}

#define N_ROWS   500000
#define DIM      256
#define KCOMM    128
#define NTILES   31250          // 500000 / 16
#define WPB      8              // waves per block (256 threads, wave32)
#define NBLOCKS  512

__global__ __launch_bounds__(256) void assign_wmma_kernel(
    const float* __restrict__ seq,   // [N, 256]
    const float* __restrict__ uin,   // [N, 128]
    const float* __restrict__ cemb,  // [128, 256]
    float* __restrict__ out)         // [32768 passthrough][N*128 assignmat]
{
    // 64 KB LDS: bf16 C packed in WMMA-B layout.
    // Region (t,c) = 1024 B: lane half0 @ +lane*16, half1 @ +512+lane*16
    // (16 B interleave => ds_load_b128 at the 2-row minimum bank cost).
    __shared__ __align__(16) unsigned char smem[65536];

    const int tid  = threadIdx.x;
    const int lane = tid & 31;
    const int wv   = tid >> 5;
    const int hi   = lane >> 4;      // K-half selector (wave32 WMMA layouts)
    const int lm   = lane & 15;

    // ---- phase 1: c2[k] = sum_d C[k][d]^2 (f32), staged in LDS then regs ----
    float* c2p = (float*)smem;              // 256 partials
    float* c2f = (float*)(smem + 1024);     // 128 finals
    {
        const int k = tid >> 1, half = tid & 1;
        const float* row = cemb + k * DIM + half * 128;
        float s = 0.f;
        #pragma unroll 4
        for (int d = 0; d < 128; ++d) { float v = row[d]; s = fmaf(v, v, s); }
        c2p[tid] = s;
    }
    __syncthreads();
    if (tid < KCOMM) c2f[tid] = c2p[2 * tid] + c2p[2 * tid + 1];
    __syncthreads();
    float c2reg[8];
    #pragma unroll
    for (int t = 0; t < 8; ++t) c2reg[t] = c2f[t * 16 + lm];
    __syncthreads();   // everyone holds c2 in regs; LDS free to be overwritten

    // ---- phase 2: pack C -> bf16 WMMA-B fragments in LDS -------------------
    // Lane l of tile (t,c) needs C[16t + (l&15)][32c + 16*(l>>4) + e], e=0..15.
    for (int o = tid; o < 4096; o += 256) {      // 4096 octets of 8 floats
        const int k = o >> 5, j = o & 31;        // row k, octet j (d = 8j..)
        const float4* src = (const float4*)(cemb + k * DIM + j * 8);
        float4 a = src[0], b = src[1];
        BFrag p;                                  // use low 4 dwords only
        bf_pair(p, 0, a.x, a.y); bf_pair(p, 1, a.z, a.w);
        bf_pair(p, 2, b.x, b.y); bf_pair(p, 3, b.z, b.w);
        uint4 q = { p.u[0], p.u[1], p.u[2], p.u[3] };
        const int region = ((k >> 4) << 3) + (j >> 2);           // t*8 + c
        const int lanei  = (k & 15) + (((j >> 1) & 1) << 4);     // n + 16*hi
        *(uint4*)(smem + region * 1024 + ((j & 1) << 9) + lanei * 16) = q;
    }
    __syncthreads();

    // ---- main loop: 16-row x 128-col tiles ---------------------------------
    const int gw = blockIdx.x * WPB + wv;
    const int nw = gridDim.x * WPB;
    for (int tile = gw; tile < NTILES; tile += nw) {
        const int rowbase = tile << 4;
        const float* xrow = seq + (size_t)(rowbase + lm) * DIM;

        v8f acc[8];
        const v8f vzero = {0.f, 0.f, 0.f, 0.f, 0.f, 0.f, 0.f, 0.f};
        #pragma unroll
        for (int t = 0; t < 8; ++t) acc[t] = vzero;
        float x2p = 0.f;

        // prefetch next tile's seq block (global_prefetch_b8 flavor)
        if (tile + nw < NTILES) {
            const char* nx = (const char*)(seq + (size_t)(tile + nw) * 16 * DIM);
            __builtin_prefetch(nx + lane * 512, 0, 1);
            __builtin_prefetch(nx + lane * 512 + 256, 0, 1);
        }

        #pragma unroll 2
        for (int c = 0; c < 8; ++c) {
            // A fragment (16x32 bf16): lane holds row (l&15),
            // K = 16*(l>>4)+[0,16) in elems 0..15
            const float4* xp = (const float4*)(xrow + c * 32 + hi * 16);
            float4 a0 = xp[0], a1 = xp[1], a2 = xp[2], a3 = xp[3];
            x2p = sumsq4(a0, x2p); x2p = sumsq4(a1, x2p);
            x2p = sumsq4(a2, x2p); x2p = sumsq4(a3, x2p);
            BFrag A;
            bf_pair(A, 0, a0.x, a0.y); bf_pair(A, 1, a0.z, a0.w);
            bf_pair(A, 2, a1.x, a1.y); bf_pair(A, 3, a1.z, a1.w);
            bf_pair(A, 4, a2.x, a2.y); bf_pair(A, 5, a2.z, a2.w);
            bf_pair(A, 6, a3.x, a3.y); bf_pair(A, 7, a3.z, a3.w);

            #pragma unroll
            for (int t = 0; t < 8; ++t) {
                const uint4* bp =
                    (const uint4*)(smem + ((t << 3) + c) * 1024 + lane * 16);
                uint4 b0 = bp[0];       // half0
                uint4 b1 = bp[32];      // half1 (+512 B)
                BFrag B;
                B.u[0] = b0.x; B.u[1] = b0.y; B.u[2] = b0.z; B.u[3] = b0.w;
                B.u[4] = b1.x; B.u[5] = b1.y; B.u[6] = b1.z; B.u[7] = b1.w;
                acc[t] = __builtin_amdgcn_wmma_f32_16x16x32_bf16(
                    false, A.v, false, B.v, (short)0, acc[t], false, false);
            }
        }

        // x2 of row (l&15): lane l + lane l^16 covered disjoint halves of d
        float x2s = x2p + __shfl_xor(x2p, 16, 32);

        // Epilogue. C/D layout: lane l holds rows m = r + 8*(l>>4),
        // cols n = 16t + (l&15). Each row spans one 16-lane half-group.
        #pragma unroll
        for (int r = 0; r < 8; ++r) {
            const int m = r + (hi << 3);
            const float x2v = __shfl(x2s, m, 32);  // lane m holds x2 of row m
            const float* urow = uin + (size_t)(rowbase + m) * KCOMM + lm;

            float z[8], e[8];
            #pragma unroll
            for (int t = 0; t < 8; ++t) {
                float dotv = acc[t][r];
                float sq   = x2v - 2.f * dotv + c2reg[t];
                float sc   = sqrtf(fmaxf(sq, 0.f));
                float uu   = urow[t * 16];
                float g    = -__logf(-__logf(uu + 1e-10f) + 1e-10f);
                z[t] = sc + g;                      // tau = 1
            }
            // argmax with first-max tie-break (like jnp.argmax)
            float zmax = z[0]; int imax = lm;
            #pragma unroll
            for (int t = 1; t < 8; ++t) {
                const int n = (t << 4) + lm;
                if (z[t] > zmax) { zmax = z[t]; imax = n; }
            }
            #pragma unroll
            for (int s = 1; s < 16; s <<= 1) {      // reduce within 16 lanes
                float oz = __shfl_xor(zmax, s, 32);
                int   oi = __shfl_xor(imax, s, 32);
                if (oz > zmax || (oz == zmax && oi < imax)) { zmax = oz; imax = oi; }
            }
            float esum = 0.f;
            #pragma unroll
            for (int t = 0; t < 8; ++t) { e[t] = __expf(z[t] - zmax); esum += e[t]; }
            #pragma unroll
            for (int s = 1; s < 16; s <<= 1) esum += __shfl_xor(esum, s, 32);
            const float rinv = 1.0f / esum;         // one divide per row

            float* orow = out + 32768 + (size_t)(rowbase + m) * KCOMM + lm;
            #pragma unroll
            for (int t = 0; t < 8; ++t) {
                float sv = e[t] * rinv;
                float hv = (((t << 4) + lm) == imax) ? 1.f : 0.f;
                orow[t * 16] = (hv + sv) - sv;      // y_hard + y_soft - sg(y_soft)
            }
        }
    }
}

extern "C" void kernel_launch(void* const* d_in, const int* in_sizes, int n_in,
                              void* d_out, int out_size, void* d_ws, size_t ws_size,
                              hipStream_t stream) {
    (void)in_sizes; (void)n_in; (void)out_size; (void)d_ws; (void)ws_size;
    const float* seq = (const float*)d_in[0];
    const float* uin = (const float*)d_in[1];
    const float* ce  = (const float*)d_in[2];
    float* out = (float*)d_out;
    // output[0] = community_embed passthrough (128*256 f32)
    hipMemcpyAsync(out, ce, (size_t)KCOMM * DIM * sizeof(float),
                   hipMemcpyDeviceToDevice, stream);
    assign_wmma_kernel<<<NBLOCKS, 256, 0, stream>>>(seq, uin, ce, out);
}
